// LinearAndCEL_11879879543617
// MI455X (gfx1250) — matrix operand order repaired and checked by reading, not compiled, not run on hardware
//
#include <hip/hip_runtime.h>
#include <hip/hip_bf16.h>

// Problem constants (from the reference)
#define R_ROWS 8192        // B*T
#define CDIM   1024        // C
#define VDIM   50257       // V
#define VPAD   50432       // V padded to multiple of 256 (197*256)
#define BM     32          // rows per workgroup (2 WMMA M-subtiles per wave)
#define BN     256         // columns per N-iteration (8 waves x 2 groups x 16)
#define NB_ITERS (VPAD / BN)   // 197
#define MTILES   (R_ROWS / BM) // 256

typedef __bf16 v16bf __attribute__((ext_vector_type(16)));
typedef __bf16 v8bf  __attribute__((ext_vector_type(8)));
typedef float  v8f   __attribute__((ext_vector_type(8)));

// ---------------------------------------------------------------------------
// f32 -> bf16 conversion of x (exact size, multiple of 4)
// ---------------------------------------------------------------------------
__global__ void cvt_x_kernel(const float* __restrict__ x,
                             __hip_bfloat16* __restrict__ o) {
    size_t i = ((size_t)blockIdx.x * 256 + threadIdx.x) * 4;
    float4 v = *(const float4*)(x + i);
    o[i + 0] = __float2bfloat16(v.x);
    o[i + 1] = __float2bfloat16(v.y);
    o[i + 2] = __float2bfloat16(v.z);
    o[i + 3] = __float2bfloat16(v.w);
}

// ---------------------------------------------------------------------------
// f32 -> bf16 conversion of W, zero-padded rows V..VPAD-1
// ---------------------------------------------------------------------------
__global__ void cvt_w_kernel(const float* __restrict__ w,
                             __hip_bfloat16* __restrict__ o) {
    size_t i = ((size_t)blockIdx.x * 256 + threadIdx.x) * 4;
    const size_t limit = (size_t)VDIM * CDIM;   // multiple of 4
    if (i + 4 <= limit) {
        float4 v = *(const float4*)(w + i);
        o[i + 0] = __float2bfloat16(v.x);
        o[i + 1] = __float2bfloat16(v.y);
        o[i + 2] = __float2bfloat16(v.z);
        o[i + 3] = __float2bfloat16(v.w);
    } else {  // pure padding region
        o[i + 0] = __float2bfloat16(0.f);
        o[i + 1] = __float2bfloat16(0.f);
        o[i + 2] = __float2bfloat16(0.f);
        o[i + 3] = __float2bfloat16(0.f);
    }
}

// Online log-sum-exp update with a single transcendental:
//   e = exp(-|v-m|);  v>m ? (s = s*e + 1, m = v) : (s = s + e)
__device__ __forceinline__ void lse_update(float& m, float& s, float v) {
    float e = __expf(-fabsf(v - m));
    bool gt = v > m;
    s = gt ? __builtin_fmaf(s, e, 1.0f) : (s + e);
    m = fmaxf(m, v);
}

// ---------------------------------------------------------------------------
// Fused GEMM (x @ W^T + b) + online log-sum-exp + label-logit gather.
// One workgroup = 32 rows x full V. 8 waves; each wave owns a 32x32 register
// tile per N-iteration (2 M-subtiles x 2 N-groups, 4 WMMAs per k-step).
// ---------------------------------------------------------------------------
__global__ __launch_bounds__(256) void fused_lce_kernel(
    const __hip_bfloat16* __restrict__ xbf,
    const __hip_bfloat16* __restrict__ wbf,
    const int*            __restrict__ y,
    const float*          __restrict__ bias,
    float*                __restrict__ partial) {
    // 64 KB x-tile; re-used (after barrier) as the cross-wave merge buffer.
    __shared__ __align__(16) unsigned short xs[BM * CDIM];

    const int tid    = threadIdx.x;
    const int wave   = tid >> 5;
    const int lane   = tid & 31;
    const int laneN  = lane & 15;     // N index (B col) and A row within subtile
    const int laneHi = lane >> 4;     // selects K-half (A) / K-range (B), M-half (C)
    const int mtile  = blockIdx.x;
    const int row0   = mtile * BM;

    // ---- stage x tile (BM x CDIM bf16) into LDS via async global->LDS ----
    {
        const unsigned long long src = (unsigned long long)(xbf + (size_t)row0 * CDIM);
        for (int c = tid; c < BM * CDIM / 8; c += 256) {
            unsigned ldsOff = (unsigned)(size_t)(&xs[c * 8]);  // LDS byte offset
            unsigned gOff   = (unsigned)(c * 16);              // global byte offset
            asm volatile("global_load_async_to_lds_b128 %0, %1, %2"
                         :: "v"(ldsOff), "v"(gOff), "s"(src)
                         : "memory");
        }
        asm volatile("s_wait_asynccnt 0" ::: "memory");
    }
    __syncthreads();

    // ---- per-slot labels: slot (s,i) -> row s*16 + i + 8*laneHi ----
    int lbl[2][8];
#pragma unroll
    for (int s = 0; s < 2; ++s)
#pragma unroll
        for (int i = 0; i < 8; ++i)
            lbl[s][i] = y[row0 + s * 16 + 8 * laneHi + i];

    // running per-slot stats (own column subset of the row)
    float mrun[2][8], srun[2][8], lrun[2][8];
#pragma unroll
    for (int s = 0; s < 2; ++s)
#pragma unroll
        for (int i = 0; i < 8; ++i) {
            mrun[s][i] = -3.0e38f;
            srun[s][i] = 0.f;
            lrun[s][i] = 0.f;
        }

    const int ksel = laneHi * 16;   // A-matrix K-offset for this lane half
    const int bsel = laneHi * 16;   // B-matrix K-offset for this lane half

    for (int nb = 0; nb < NB_ITERS; ++nb) {
        const int  col0   = nb * BN + wave * 16 + laneN;        // group 0
        const int  col1   = col0 + 128;                         // group 1
        const bool valid0 = (col0 < VDIM);
        const bool valid1 = (col1 < VDIM);
        const float b0 = valid0 ? bias[col0] : 0.f;
        const float b1 = valid1 ? bias[col1] : 0.f;
        const __hip_bfloat16* wp0 = wbf + (size_t)col0 * CDIM;
        const __hip_bfloat16* wp1 = wbf + (size_t)col1 * CDIM;

        // prefetch next N-iteration's W rows into the cache hierarchy
        if (col1 + BN < VPAD) {
            __builtin_prefetch(wp0 + (size_t)BN * CDIM, 0, 1);
            __builtin_prefetch(wp1 + (size_t)BN * CDIM, 0, 1);
        }

        v8f acc00 = {}, acc01 = {}, acc10 = {}, acc11 = {};
#pragma unroll 2
        for (int k = 0; k < CDIM; k += 32) {
            const int k0 = k + ksel;
            // A fragments from LDS (contiguous layout: K in {k0..k0+15})
            v8bf a0lo = *(const v8bf*)(&xs[(laneN)      * CDIM + k0]);
            v8bf a0hi = *(const v8bf*)(&xs[(laneN)      * CDIM + k0 + 8]);
            v8bf a1lo = *(const v8bf*)(&xs[(16 + laneN) * CDIM + k0]);
            v8bf a1hi = *(const v8bf*)(&xs[(16 + laneN) * CDIM + k0 + 8]);
            v16bf af0 = __builtin_shufflevector(a0lo, a0hi, 0, 1, 2, 3, 4, 5, 6, 7,
                                                8, 9, 10, 11, 12, 13, 14, 15);
            v16bf af1 = __builtin_shufflevector(a1lo, a1hi, 0, 1, 2, 3, 4, 5, 6, 7,
                                                8, 9, 10, 11, 12, 13, 14, 15);
            // B fragments (contiguous layout, same as A: lane half selects contiguous K range)
            v16bf bf0 = *(const v16bf*)(wp0 + k + bsel);
            v16bf bf1 = *(const v16bf*)(wp1 + k + bsel);

            acc00 = __builtin_amdgcn_wmma_f32_16x16x32_bf16(
                false, af0, false, bf0, (short)0, acc00, false, false);
            acc10 = __builtin_amdgcn_wmma_f32_16x16x32_bf16(
                false, af1, false, bf0, (short)0, acc10, false, false);
            acc01 = __builtin_amdgcn_wmma_f32_16x16x32_bf16(
                false, af0, false, bf1, (short)0, acc01, false, false);
            acc11 = __builtin_amdgcn_wmma_f32_16x16x32_bf16(
                false, af1, false, bf1, (short)0, acc11, false, false);
        }

        // online softmax-stat update; C/D slot (i,lane) is M=i+8*laneHi, N=laneN
#pragma unroll
        for (int i = 0; i < 8; ++i) {
            float v00 = valid0 ? (acc00[i] + b0) : -__builtin_inff();
            float v01 = valid1 ? (acc01[i] + b1) : -__builtin_inff();
            float v10 = valid0 ? (acc10[i] + b0) : -__builtin_inff();
            float v11 = valid1 ? (acc11[i] + b1) : -__builtin_inff();

            lse_update(mrun[0][i], srun[0][i], v00);
            lse_update(mrun[0][i], srun[0][i], v01);
            lse_update(mrun[1][i], srun[1][i], v10);
            lse_update(mrun[1][i], srun[1][i], v11);

            if (col0 == lbl[0][i]) lrun[0][i] += v00;
            if (col1 == lbl[0][i]) lrun[0][i] += v01;
            if (col0 == lbl[1][i]) lrun[1][i] += v10;
            if (col1 == lbl[1][i]) lrun[1][i] += v11;
        }
    }

    // ---- merge across the 16 N-lanes of each half-wave (butterfly) ----
    float wm[2][8], wsum[2][8], wlab[2][8];
#pragma unroll
    for (int s = 0; s < 2; ++s)
#pragma unroll
        for (int i = 0; i < 8; ++i) {
            float m = mrun[s][i], ss = srun[s][i], ll = lrun[s][i];
            for (int d = 1; d < 16; d <<= 1) {
                float mo = __shfl_xor(m, d, 32);
                float so = __shfl_xor(ss, d, 32);
                float lo = __shfl_xor(ll, d, 32);
                float nm = fmaxf(m, mo);
                ss = ss * __expf(m - nm) + so * __expf(mo - nm);
                m  = nm;
                ll += lo;
            }
            wm[s][i] = m; wsum[s][i] = ss; wlab[s][i] = ll;
        }

    // all waves done with xs: re-use it as the cross-wave merge buffer
    __syncthreads();
    float* red_m = (float*)xs;          // [8][BM]
    float* red_s = red_m + 8 * BM;
    float* red_l = red_s + 8 * BM;
    if (laneN == 0) {
#pragma unroll
        for (int s = 0; s < 2; ++s)
#pragma unroll
            for (int i = 0; i < 8; ++i) {
                int r = s * 16 + 8 * laneHi + i;
                red_m[wave * BM + r] = wm[s][i];
                red_s[wave * BM + r] = wsum[s][i];
                red_l[wave * BM + r] = wlab[s][i];
            }
    }
    __syncthreads();

    // ---- merge across the 8 waves; one thread per row, then block sum ----
    if (tid < BM) {
        const int r = tid;
        float m = red_m[r], ss = red_s[r], ll = red_l[r];
        for (int w = 1; w < 8; ++w) {
            float mo = red_m[w * BM + r];
            float so = red_s[w * BM + r];
            float nm = fmaxf(m, mo);
            ss = ss * __expf(m - nm) + so * __expf(mo - nm);
            m  = nm;
            ll += red_l[w * BM + r];
        }
        float loss = m + __logf(ss) - ll;     // logZ - logit[label]
        for (int d = 1; d < 32; d <<= 1) loss += __shfl_xor(loss, d, 32);
        if (tid == 0) partial[mtile] = loss;
    }
}

// ---------------------------------------------------------------------------
// Final reduction: 256 partials -> mean NLL
// ---------------------------------------------------------------------------
__global__ void reduce_kernel(const float* __restrict__ partial,
                              float* __restrict__ out) {
    __shared__ float sm[256];
    sm[threadIdx.x] = partial[threadIdx.x];
    __syncthreads();
    for (int s = 128; s > 0; s >>= 1) {
        if (threadIdx.x < s) sm[threadIdx.x] += sm[threadIdx.x + s];
        __syncthreads();
    }
    if (threadIdx.x == 0) out[0] = sm[0] * (1.0f / (float)R_ROWS);
}

// ---------------------------------------------------------------------------
// Launch: cvt x, cvt W (bf16, L2-resident), fused GEMM+CE, reduce.
// Workspace: xbf (16 MB) | wbf (98.5 MB) | partial (1 KB)  ~= 120 MB
// ---------------------------------------------------------------------------
extern "C" void kernel_launch(void* const* d_in, const int* in_sizes, int n_in,
                              void* d_out, int out_size, void* d_ws, size_t ws_size,
                              hipStream_t stream) {
    const float* x = (const float*)d_in[0];
    const int*   y = (const int*)d_in[1];
    const float* W = (const float*)d_in[2];
    const float* b = (const float*)d_in[3];

    char* ws = (char*)d_ws;
    __hip_bfloat16* xbf = (__hip_bfloat16*)ws;
    __hip_bfloat16* wbf = (__hip_bfloat16*)(ws + (size_t)R_ROWS * CDIM * 2);
    float* partial = (float*)(ws + (size_t)R_ROWS * CDIM * 2 + (size_t)VPAD * CDIM * 2);

    cvt_x_kernel<<<(R_ROWS * CDIM) / 4 / 256, 256, 0, stream>>>(x, xbf);
    cvt_w_kernel<<<((size_t)VPAD * CDIM) / 4 / 256, 256, 0, stream>>>(W, wbf);
    fused_lce_kernel<<<MTILES, 256, 0, stream>>>(xbf, wbf, y, b, partial);
    reduce_kernel<<<1, 256, 0, stream>>>(partial, (float*)d_out);
}
